// ToMeSABlock_11458972745806
// MI455X (gfx1250) — compile-verified, matrix operand order checked by reading, not run on hardware
//
#include <hip/hip_runtime.h>
#include <hip/hip_bf16.h>
#include <math.h>

typedef __attribute__((ext_vector_type(16))) _Float16 v16h;
typedef __attribute__((ext_vector_type(8)))  float    v8f;

#define WMMA16(a,b,c) __builtin_amdgcn_wmma_f32_16x16x32_f16(false,(a),false,(b),(short)0,(c),false,false)

namespace tome {
constexpr int Bc = 4, Dd = 512, Tt = 16, Hh = 8, HD = 64;
constexpr int Cc = 512;
constexpr int Nn = Bc * Tt;          // 64
constexpr int TQ = Cc;               // 512 tokens per "batch"
constexpr int NTOK = Nn * TQ;        // 32768 rows
constexpr int Rr = 128;
constexpr int TNEW = TQ - Rr;        // 384
constexpr int NM = Nn * TNEW;        // 24576 rows after merge
constexpr int DFF = 4 * Dd;          // 2048
}

// ---------------------------------------------------------------------------
// WMMA fragment loader for row-major f16 tiles.
// A-frag 16x32 (MxK):  lane L holds row (L&15); per-lane data = 8 contiguous
// halfs at k = ((L>>4)*8) and 8 contiguous halfs at k+16 -> two b128 loads.
// B-frag 32x16 (KxN) from a row-major [N,K] matrix uses the identical pattern
// with "row" = output column. C/D frag: lane L, vgpr v -> (m=(L>>4)*8+v, n=L&15).
// ---------------------------------------------------------------------------
__device__ __forceinline__ v16h load_frag(const _Float16* base, int ld, int lane) {
  const _Float16* p = base + (size_t)(lane & 15) * ld + ((lane >> 4) << 3);
  v16h r;
  float4* q = reinterpret_cast<float4*>(&r);
  q[0] = *reinterpret_cast<const float4*>(p);
  q[1] = *reinterpret_cast<const float4*>(p + 16);
  return r;
}

// ---------------------------------------------------------------------------
// Core: one wave computes a 32x64 output tile = 2x4 WMMA fragments.
// Register budget (fits 128-VGPR allocation, no spills): 8 x v8f acc (64) +
// 2 x v16h A (16) + 1-2 x v16h B (8-16) + addressing (~12).
// Per 32-wide k-step: 4 A-loads + 8 B-loads feed 8 WMMAs; each B fragment is
// followed by 2 independent WMMAs. K-loop chunked: unrolled 128-wide inner,
// non-unrolled outer (code size). A pre-offset to row m0, W to row n0, ld=K.
// ---------------------------------------------------------------------------
template <int K>
__device__ __forceinline__ void gemm_core32(const _Float16* __restrict__ A,
                                            const _Float16* __restrict__ W,
                                            int lane, v8f c[2][4]) {
  constexpr int CH = 128;
  const int arow = (lane & 15);
#pragma unroll 1
  for (int kk = 0; kk < K; kk += CH) {
    if (kk + CH < K) {
      // prefetch next A chunk (streaming operand; weights stay L2-resident)
      __builtin_prefetch(A + (size_t)arow * K + kk + CH, 0, 1);
      __builtin_prefetch(A + (size_t)(16 + arow) * K + kk + CH, 0, 1);
    }
#pragma unroll
    for (int k0 = 0; k0 < CH; k0 += 32) {
      v16h a0 = load_frag(A + kk + k0, K, lane);
      v16h a1 = load_frag(A + (size_t)16 * K + kk + k0, K, lane);
#pragma unroll
      for (int nf = 0; nf < 4; ++nf) {
        v16h b = load_frag(W + (size_t)(nf * 16) * K + kk + k0, K, lane);
        c[0][nf] = WMMA16(a0, b, c[0][nf]);
        c[1][nf] = WMMA16(a1, b, c[1][nf]);
      }
    }
  }
}

// ---------------------------------------------------------------------------
// f32 -> f16 conversion (weights)
// ---------------------------------------------------------------------------
__global__ void f32_to_f16_kernel(const float* __restrict__ src,
                                  _Float16* __restrict__ dst, int count) {
  int i = blockIdx.x * blockDim.x + threadIdx.x;
  int stride = gridDim.x * blockDim.x;
  for (; i < count; i += stride) dst[i] = (_Float16)src[i];
}

// ---------------------------------------------------------------------------
// Fused gather-transpose x(B,C,D,T) -> h(n,t,D) + LayerNorm1 -> y16
// ---------------------------------------------------------------------------
__global__ void prep_ln_kernel(const float* __restrict__ x, float* __restrict__ h,
                               _Float16* __restrict__ y16,
                               const float* __restrict__ w, const float* __restrict__ bb) {
  using namespace tome;
  int row = blockIdx.x;              // row = n*512 + c
  int tid = threadIdx.x;
  int n = row >> 9, c = row & 511;
  int b = n >> 4, t = n & 15;        // n = b*T + t
  const float* xp = x + ((size_t)(b * Cc + c) * Dd) * Tt + t;
  float x0 = xp[(size_t)tid * Tt];
  float x1 = xp[(size_t)(tid + 256) * Tt];

  __shared__ float red[256];
  red[tid] = x0 + x1;
  __syncthreads();
  for (int off = 128; off > 0; off >>= 1) {
    if (tid < off) red[tid] += red[tid + off];
    __syncthreads();
  }
  float mean = red[0] * (1.0f / 512.0f);
  __syncthreads();
  float d0 = x0 - mean, d1 = x1 - mean;
  red[tid] = d0 * d0 + d1 * d1;
  __syncthreads();
  for (int off = 128; off > 0; off >>= 1) {
    if (tid < off) red[tid] += red[tid + off];
    __syncthreads();
  }
  float rstd = rsqrtf(red[0] * (1.0f / 512.0f) + 1e-6f);

  size_t o = (size_t)row * Dd;
  h[o + tid] = x0;
  h[o + tid + 256] = x1;
  y16[o + tid] = (_Float16)(d0 * rstd * w[tid] + bb[tid]);
  y16[o + tid + 256] = (_Float16)(d1 * rstd * w[tid + 256] + bb[tid + 256]);
}

// ---------------------------------------------------------------------------
// QKV GEMM: [32768,512] x [1536,512]^T, 32x64 wave tiles, scatter epilogue into
// per-head q16/k16 ([n,h][tok][hd]) and transposed vT16 ([n,h][hd][tok]).
// ---------------------------------------------------------------------------
__global__ void __launch_bounds__(256, 2)
gemm_qkv_kernel(const _Float16* __restrict__ A, const _Float16* __restrict__ W,
                const float* __restrict__ bias,
                _Float16* __restrict__ q16, _Float16* __restrict__ k16,
                _Float16* __restrict__ vT16, int tilesTotal) {
  using namespace tome;
  int lane = threadIdx.x & 31, wave = threadIdx.x >> 5;
  int tile = blockIdx.x * 8 + wave;
  if (tile >= tilesTotal) return;
  const int tilesN = (3 * Dd) / 64;  // 24
  int m0 = (tile / tilesN) * 32, n0 = (tile % tilesN) * 64;

  v8f c[2][4] = {};
  gemm_core32<Dd>(A + (size_t)m0 * Dd, W + (size_t)n0 * Dd, lane, c);

  int rbase = (lane >> 4) * 8;
  int cl = lane & 15;
  int nbatch = m0 >> 9;              // uniform per tile (32 | 512)
#pragma unroll
  for (int nf = 0; nf < 4; ++nf) {
    int colbase = n0 + nf * 16;
    int p = colbase >> 9;            // 0=q,1=k,2=v
    int head = (colbase >> 6) & 7;
    int e = (colbase & 63) + cl;
    float bv = bias[colbase + cl];
#pragma unroll
    for (int mf = 0; mf < 2; ++mf) {
#pragma unroll
      for (int v = 0; v < 8; ++v) {
        int row = m0 + mf * 16 + rbase + v;
        int tok = row & 511;
        _Float16 hv = (_Float16)(c[mf][nf][v] + bv);
        size_t qkidx = (((size_t)(nbatch * Hh + head) * TQ) + tok) * HD + e;
        if (p == 0)      q16[qkidx] = hv;
        else if (p == 1) k16[qkidx] = hv;
        else             vT16[(((size_t)(nbatch * Hh + head) * HD) + e) * TQ + tok] = hv;
      }
    }
  }
}

// ---------------------------------------------------------------------------
// Generic GEMM epilogues: GELU->f16 (fc1) and add-into-f32-residual (proj/fc2)
// ---------------------------------------------------------------------------
template <int NC, int K>
__global__ void __launch_bounds__(256, 2)
gemm_gelu_kernel(const _Float16* __restrict__ A, const _Float16* __restrict__ W,
                 const float* __restrict__ bias, _Float16* __restrict__ out16,
                 int tilesTotal) {
  int lane = threadIdx.x & 31, wave = threadIdx.x >> 5;
  int tile = blockIdx.x * 8 + wave;
  if (tile >= tilesTotal) return;
  constexpr int tilesN = NC / 64;
  int m0 = (tile / tilesN) * 32, n0 = (tile % tilesN) * 64;
  v8f c[2][4] = {};
  gemm_core32<K>(A + (size_t)m0 * K, W + (size_t)n0 * K, lane, c);
  int rbase = (lane >> 4) * 8, cl = lane & 15;
#pragma unroll
  for (int nf = 0; nf < 4; ++nf) {
    int col = n0 + nf * 16 + cl;
    float bv = bias[col];
#pragma unroll
    for (int mf = 0; mf < 2; ++mf) {
#pragma unroll
      for (int v = 0; v < 8; ++v) {
        int row = m0 + mf * 16 + rbase + v;
        float val = c[mf][nf][v] + bv;
        val = 0.5f * val * (1.0f + erff(val * 0.70710678118f));   // exact GELU
        out16[(size_t)row * NC + col] = (_Float16)val;
      }
    }
  }
}

template <int NC, int K>
__global__ void __launch_bounds__(256, 2)
gemm_addto_kernel(const _Float16* __restrict__ A, const _Float16* __restrict__ W,
                  const float* __restrict__ bias, float* __restrict__ outf,
                  int tilesTotal) {
  int lane = threadIdx.x & 31, wave = threadIdx.x >> 5;
  int tile = blockIdx.x * 8 + wave;
  if (tile >= tilesTotal) return;
  constexpr int tilesN = NC / 64;
  int m0 = (tile / tilesN) * 32, n0 = (tile % tilesN) * 64;
  v8f c[2][4] = {};
  gemm_core32<K>(A + (size_t)m0 * K, W + (size_t)n0 * K, lane, c);
  int rbase = (lane >> 4) * 8, cl = lane & 15;
#pragma unroll
  for (int nf = 0; nf < 4; ++nf) {
    int col = n0 + nf * 16 + cl;
    float bv = bias[col];
#pragma unroll
    for (int mf = 0; mf < 2; ++mf) {
#pragma unroll
      for (int v = 0; v < 8; ++v) {
        int row = m0 + mf * 16 + rbase + v;
        outf[(size_t)row * NC + col] += c[mf][nf][v] + bv;
      }
    }
  }
}

// ---------------------------------------------------------------------------
// Flash attention: grid (N*H, TQ/128), block 256 = 8 waves, wave = 16 q-rows.
// 32 keys per iteration: QK^T via 4 WMMA, online softmax (shfl butterflies),
// P staged via LDS into A-frag layout, PV via 4 WMMA (V pre-transposed).
// ---------------------------------------------------------------------------
__global__ void __launch_bounds__(256, 2)
attn_kernel(const _Float16* __restrict__ q16, const _Float16* __restrict__ k16,
            const _Float16* __restrict__ vT16, _Float16* __restrict__ o16) {
  using namespace tome;
  int lane = threadIdx.x & 31, wave = threadIdx.x >> 5;
  int nh = blockIdx.x;
  int n = nh >> 3, head = nh & 7;
  int tq0 = blockIdx.y * 128 + wave * 16;

  const _Float16* Q  = q16  + ((size_t)nh * TQ + tq0) * HD;
  const _Float16* Kp = k16  + (size_t)nh * TQ * HD;
  const _Float16* Vt = vT16 + (size_t)nh * HD * TQ;

  v16h qa0 = load_frag(Q, HD, lane);
  v16h qa1 = load_frag(Q + 32, HD, lane);

  float mi[8], li[8];
#pragma unroll
  for (int v = 0; v < 8; ++v) { mi[v] = -1e30f; li[v] = 0.0f; }
  v8f oacc[4] = {};

  __shared__ __align__(16) _Float16 Pbuf[8][16 * 32];
  _Float16* Pw = &Pbuf[wave][0];
  int cl = lane & 15;

#pragma unroll 1
  for (int kt = 0; kt < TQ; kt += 32) {
    v8f s0 = {}, s1 = {};
    {
      v16h b0 = load_frag(Kp + (size_t)kt * HD, HD, lane);
      v16h b1 = load_frag(Kp + (size_t)kt * HD + 32, HD, lane);
      s0 = WMMA16(qa0, b0, s0);
      s0 = WMMA16(qa1, b1, s0);
      v16h b2 = load_frag(Kp + (size_t)(kt + 16) * HD, HD, lane);
      v16h b3 = load_frag(Kp + (size_t)(kt + 16) * HD + 32, HD, lane);
      s1 = WMMA16(qa0, b2, s1);
      s1 = WMMA16(qa1, b3, s1);
    }
#pragma unroll
    for (int v = 0; v < 8; ++v) {
      float a0 = s0[v] * 0.125f;    // hd^-0.5 = 1/8
      float a1 = s1[v] * 0.125f;
      float x = fmaxf(a0, a1);
      for (int off = 1; off < 16; off <<= 1) x = fmaxf(x, __shfl_xor(x, off, 32));
      float newm = fmaxf(mi[v], x);
      float corr = __expf(mi[v] - newm);
      float p0 = __expf(a0 - newm), p1 = __expf(a1 - newm);
      float rs = p0 + p1;
      for (int off = 1; off < 16; off <<= 1) rs += __shfl_xor(rs, off, 32);
      li[v] = li[v] * corr + rs;
      mi[v] = newm;
      int m = (lane >> 4) * 8 + v;
      Pw[m * 32 + cl]      = (_Float16)p0;
      Pw[m * 32 + 16 + cl] = (_Float16)p1;
#pragma unroll
      for (int t = 0; t < 4; ++t) oacc[t][v] *= corr;
    }
    __syncthreads();
    v16h pf = load_frag(Pw, 32, lane);
#pragma unroll
    for (int t = 0; t < 4; ++t) {
      v16h bf = load_frag(Vt + (size_t)(t * 16) * TQ + kt, TQ, lane);
      oacc[t] = WMMA16(pf, bf, oacc[t]);
    }
    __syncthreads();
  }

#pragma unroll
  for (int t = 0; t < 4; ++t)
#pragma unroll
    for (int v = 0; v < 8; ++v) {
      int row = tq0 + (lane >> 4) * 8 + v;
      float val = oacc[t][v] / li[v];
      o16[((size_t)(n * TQ) + row) * Dd + head * HD + t * 16 + cl] = (_Float16)val;
    }
}

// ---------------------------------------------------------------------------
// metric = mean(k over heads), then normalized:  mnorm (n, 512, 64)
// ---------------------------------------------------------------------------
__global__ void metric_kernel(const _Float16* __restrict__ k16, float* __restrict__ mnorm) {
  using namespace tome;
  int id = blockIdx.x;               // n*512 + tok
  int n = id >> 9, tok = id & 511;
  int e = threadIdx.x;               // 64 threads
  float s = 0.0f;
#pragma unroll
  for (int hh = 0; hh < Hh; ++hh)
    s += (float)k16[(((size_t)(n * Hh + hh) * TQ) + tok) * HD + e];
  s *= 0.125f;
  __shared__ float red[64];
  red[e] = s * s;
  __syncthreads();
  for (int off = 32; off > 0; off >>= 1) {
    if (e < off) red[e] += red[e + off];
    __syncthreads();
  }
  mnorm[(size_t)id * HD + e] = s * rsqrtf(red[0]);
}

// ---------------------------------------------------------------------------
// scores[i,j] = a_i . b_j ; per (n,i) block computes row max + argmax
// ---------------------------------------------------------------------------
__global__ void scores_kernel(const float* __restrict__ mnorm,
                              float* __restrict__ node_max, int* __restrict__ node_idx) {
  using namespace tome;
  int id = blockIdx.x;               // n*256 + i
  int n = id >> 8, i = id & 255;
  int tid = threadIdx.x;             // 256
  __shared__ float av[64];
  if (tid < 64) av[tid] = mnorm[((size_t)n * TQ + 2 * i) * HD + tid];
  __syncthreads();
  const float* bj = mnorm + ((size_t)n * TQ + 2 * tid + 1) * HD;
  float dot = 0.0f;
#pragma unroll
  for (int e = 0; e < 64; ++e) dot += av[e] * bj[e];

  __shared__ float rv[256];
  __shared__ int ri[256];
  rv[tid] = dot; ri[tid] = tid;
  __syncthreads();
  for (int off = 128; off > 0; off >>= 1) {
    if (tid < off) {
      if (rv[tid + off] > rv[tid] ||
          (rv[tid + off] == rv[tid] && ri[tid + off] < ri[tid])) {
        rv[tid] = rv[tid + off]; ri[tid] = ri[tid + off];
      }
    }
    __syncthreads();
  }
  if (tid == 0) { node_max[id] = rv[0]; node_idx[id] = ri[0]; }
}

// ---------------------------------------------------------------------------
// per-n bitonic sort of node_max (desc, tie -> lower idx first = stable argsort)
// ---------------------------------------------------------------------------
__global__ void sort_kernel(const float* __restrict__ node_max, const int* __restrict__ node_idx,
                            int* __restrict__ src_idx, int* __restrict__ dst_idx,
                            int* __restrict__ unm_idx) {
  using namespace tome;
  int n = blockIdx.x, tid = threadIdx.x;
  __shared__ float v[256];
  __shared__ int ix[256];
  v[tid] = node_max[n * 256 + tid];
  ix[tid] = tid;
  __syncthreads();
  for (int k = 2; k <= 256; k <<= 1) {
    for (int j = k >> 1; j > 0; j >>= 1) {
      int partner = tid ^ j;
      if (partner > tid) {
        float va = v[tid], vb = v[partner];
        int ia = ix[tid], ib = ix[partner];
        bool aFirst = (va > vb) || (va == vb && ia < ib);
        bool dirDesc = ((tid & k) == 0);
        if (dirDesc ? !aFirst : aFirst) {
          v[tid] = vb; v[partner] = va;
          ix[tid] = ib; ix[partner] = ia;
        }
      }
      __syncthreads();
    }
  }
  if (tid < Rr) {
    int s = ix[tid];
    src_idx[n * Rr + tid] = s;
    dst_idx[n * Rr + tid] = node_idx[n * 256 + s];
    unm_idx[n * Rr + tid] = ix[Rr + tid];
  }
}

// ---------------------------------------------------------------------------
// merge-sum: out = concat(unm(128), dst(256)); serialized adds -> deterministic
// ---------------------------------------------------------------------------
__global__ void merge_kernel(const float* __restrict__ h, float* __restrict__ hm,
                             const int* __restrict__ src_idx, const int* __restrict__ dst_idx,
                             const int* __restrict__ unm_idx) {
  using namespace tome;
  int n = blockIdx.x, tid = threadIdx.x;   // 64 blocks, 256 threads
  const float* hin = h + (size_t)n * TQ * Dd;
  float* out = hm + (size_t)n * TNEW * Dd;

  for (int u = 0; u < 128; ++u) {
    int stok = 2 * unm_idx[n * Rr + u];
    for (int d = tid; d < Dd; d += 256) out[(size_t)u * Dd + d] = hin[(size_t)stok * Dd + d];
  }
  for (int j = 0; j < 256; ++j) {
    for (int d = tid; d < Dd; d += 256)
      out[(size_t)(128 + j) * Dd + d] = hin[(size_t)(2 * j + 1) * Dd + d];
  }
  __shared__ int cnt[256];
  cnt[tid] = 1;
  __syncthreads();
  if (tid == 0)
    for (int s = 0; s < Rr; ++s) cnt[dst_idx[n * Rr + s]]++;
  __syncthreads();
  // serial adds: each thread owns fixed columns -> no races, deterministic
  for (int s = 0; s < Rr; ++s) {
    int sd = dst_idx[n * Rr + s];
    int st = 2 * src_idx[n * Rr + s];
    for (int d = tid; d < Dd; d += 256)
      out[(size_t)(128 + sd) * Dd + d] += hin[(size_t)st * Dd + d];
  }
  for (int j = 0; j < 256; ++j) {
    float inv = 1.0f / (float)cnt[j];
    for (int d = tid; d < Dd; d += 256) out[(size_t)(128 + j) * Dd + d] *= inv;
  }
}

// ---------------------------------------------------------------------------
// LayerNorm2 on merged rows -> y2_16
// ---------------------------------------------------------------------------
__global__ void ln2_kernel(const float* __restrict__ in, _Float16* __restrict__ out,
                           const float* __restrict__ w, const float* __restrict__ bb) {
  using namespace tome;
  int row = blockIdx.x, tid = threadIdx.x;
  const float* xr = in + (size_t)row * Dd;
  float x0 = xr[tid], x1 = xr[tid + 256];
  __shared__ float red[256];
  red[tid] = x0 + x1;
  __syncthreads();
  for (int off = 128; off > 0; off >>= 1) {
    if (tid < off) red[tid] += red[tid + off];
    __syncthreads();
  }
  float mean = red[0] * (1.0f / 512.0f);
  __syncthreads();
  float d0 = x0 - mean, d1 = x1 - mean;
  red[tid] = d0 * d0 + d1 * d1;
  __syncthreads();
  for (int off = 128; off > 0; off >>= 1) {
    if (tid < off) red[tid] += red[tid + off];
    __syncthreads();
  }
  float rstd = rsqrtf(red[0] * (1.0f / 512.0f) + 1e-6f);
  size_t o = (size_t)row * Dd;
  out[o + tid] = (_Float16)(d0 * rstd * w[tid] + bb[tid]);
  out[o + tid + 256] = (_Float16)(d1 * rstd * w[tid + 256] + bb[tid + 256]);
}

// ---------------------------------------------------------------------------
// final transpose: hm (b*T+t, i, d) -> out (b, i, d, t)
// ---------------------------------------------------------------------------
__global__ void out_transpose_kernel(const float* __restrict__ hm, float* __restrict__ out) {
  using namespace tome;
  size_t total = (size_t)Bc * TNEW * Dd * Tt;
  for (size_t idx = (size_t)blockIdx.x * blockDim.x + threadIdx.x; idx < total;
       idx += (size_t)gridDim.x * blockDim.x) {
    int t = (int)(idx & 15);
    size_t r = idx >> 4;
    int d = (int)(r & 511); r >>= 9;
    int i = (int)(r % TNEW);
    int b = (int)(r / TNEW);
    out[idx] = hm[(((size_t)(b * Tt + t) * TNEW) + i) * Dd + d];
  }
}

// ---------------------------------------------------------------------------
extern "C" void kernel_launch(void* const* d_in, const int* in_sizes, int n_in,
                              void* d_out, int out_size, void* d_ws, size_t ws_size,
                              hipStream_t stream) {
  using namespace tome;
  (void)in_sizes; (void)n_in; (void)out_size; (void)ws_size;
  const float* x      = (const float*)d_in[0];
  const float* qkv_w  = (const float*)d_in[1];
  const float* qkv_b  = (const float*)d_in[2];
  const float* proj_w = (const float*)d_in[3];
  const float* proj_b = (const float*)d_in[4];
  const float* fc1_w  = (const float*)d_in[5];
  const float* fc1_b  = (const float*)d_in[6];
  const float* fc2_w  = (const float*)d_in[7];
  const float* fc2_b  = (const float*)d_in[8];
  const float* n1_w   = (const float*)d_in[9];
  const float* n1_b   = (const float*)d_in[10];
  const float* n2_w   = (const float*)d_in[11];
  const float* n2_b   = (const float*)d_in[12];
  float* out = (float*)d_out;

  char* ws = (char*)d_ws;
  size_t off = 0;
  auto alloc = [&](size_t bytes) -> char* {
    char* p = ws + off;
    off = (off + bytes + 255) & ~(size_t)255;
    return p;
  };
  _Float16* qkv_w16 = (_Float16*)alloc((size_t)3 * Dd * Dd * 2);
  _Float16* proj_w16 = (_Float16*)alloc((size_t)Dd * Dd * 2);
  _Float16* fc1_w16 = (_Float16*)alloc((size_t)DFF * Dd * 2);
  _Float16* fc2_w16 = (_Float16*)alloc((size_t)Dd * DFF * 2);
  float*    h       = (float*)alloc((size_t)NTOK * Dd * 4);
  _Float16* y16     = (_Float16*)alloc((size_t)NTOK * Dd * 2);
  _Float16* q16     = (_Float16*)alloc((size_t)NTOK * Dd * 2);
  _Float16* k16     = (_Float16*)alloc((size_t)NTOK * Dd * 2);
  _Float16* vT16    = (_Float16*)alloc((size_t)NTOK * Dd * 2);
  _Float16* o16     = (_Float16*)alloc((size_t)NTOK * Dd * 2);
  float*    mnorm   = (float*)alloc((size_t)Nn * TQ * HD * 4);
  float*    node_max = (float*)alloc((size_t)Nn * 256 * 4);
  int*      node_idx = (int*)alloc((size_t)Nn * 256 * 4);
  int*      src_idx  = (int*)alloc((size_t)Nn * Rr * 4);
  int*      dst_idx  = (int*)alloc((size_t)Nn * Rr * 4);
  int*      unm_idx  = (int*)alloc((size_t)Nn * Rr * 4);
  float*    hm       = (float*)alloc((size_t)NM * Dd * 4);
  _Float16* y2_16    = (_Float16*)alloc((size_t)NM * Dd * 2);
  _Float16* g16      = (_Float16*)alloc((size_t)NM * DFF * 2);

  // 1) weights -> f16
  f32_to_f16_kernel<<<512, 256, 0, stream>>>(qkv_w, qkv_w16, 3 * Dd * Dd);
  f32_to_f16_kernel<<<512, 256, 0, stream>>>(proj_w, proj_w16, Dd * Dd);
  f32_to_f16_kernel<<<512, 256, 0, stream>>>(fc1_w, fc1_w16, DFF * Dd);
  f32_to_f16_kernel<<<512, 256, 0, stream>>>(fc2_w, fc2_w16, Dd * DFF);

  // 2) transpose + LN1
  prep_ln_kernel<<<NTOK, 256, 0, stream>>>(x, h, y16, n1_w, n1_b);

  // 3) QKV GEMM (WMMA, 32x64 wave tiles) with per-head scatter
  {
    int tiles = (NTOK / 32) * ((3 * Dd) / 64);   // 1024*24 = 24576
    gemm_qkv_kernel<<<(tiles + 7) / 8, 256, 0, stream>>>(y16, qkv_w16, qkv_b,
                                                         q16, k16, vT16, tiles);
  }

  // 4) attention (flash, WMMA)
  attn_kernel<<<dim3(Nn * Hh, TQ / 128), 256, 0, stream>>>(q16, k16, vT16, o16);

  // 5) proj GEMM (WMMA) += into residual h
  {
    int tiles = (NTOK / 32) * (Dd / 64);         // 1024*8 = 8192
    gemm_addto_kernel<Dd, Dd><<<(tiles + 7) / 8, 256, 0, stream>>>(o16, proj_w16, proj_b,
                                                                   h, tiles);
  }

  // 6) ToMe bookkeeping
  metric_kernel<<<Nn * TQ, 64, 0, stream>>>(k16, mnorm);
  scores_kernel<<<Nn * 256, 256, 0, stream>>>(mnorm, node_max, node_idx);
  sort_kernel<<<Nn, 256, 0, stream>>>(node_max, node_idx, src_idx, dst_idx, unm_idx);
  merge_kernel<<<Nn, 256, 0, stream>>>(h, hm, src_idx, dst_idx, unm_idx);

  // 7) LN2 + MLP
  ln2_kernel<<<NM, 256, 0, stream>>>(hm, y2_16, n2_w, n2_b);
  {
    int tiles = (NM / 32) * (DFF / 64);          // 768*32 = 24576
    gemm_gelu_kernel<DFF, Dd><<<(tiles + 7) / 8, 256, 0, stream>>>(y2_16, fc1_w16, fc1_b,
                                                                   g16, tiles);
  }
  {
    int tiles = (NM / 32) * (Dd / 64);           // 768*8 = 6144
    gemm_addto_kernel<Dd, DFF><<<(tiles + 7) / 8, 256, 0, stream>>>(g16, fc2_w16, fc2_b,
                                                                    hm, tiles);
  }

  // 8) final transpose to (B, t_new, D, T)
  out_transpose_kernel<<<2048, 256, 0, stream>>>(hm, out);
}
